// TransformerLayer_40742059770322
// MI455X (gfx1250) — matrix operand order repaired and checked by reading, not compiled, not run on hardware
//
#include <hip/hip_runtime.h>
#include <hip/hip_bf16.h>
#include <math.h>

typedef __attribute__((ext_vector_type(16))) __bf16 bf16x16;
typedef __attribute__((ext_vector_type(8)))  __bf16 bf16x8;
typedef __attribute__((ext_vector_type(8)))  float   f32x8;
typedef __attribute__((ext_vector_type(4)))  int     i32x4;
typedef __attribute__((address_space(1)))    i32x4   gas_i32x4;  // global
typedef __attribute__((address_space(3)))    i32x4   las_i32x4;  // LDS

#define D_MODEL 1024
#define NHEAD   16
#define DHEAD   64
#define FFDIM   4096
#define SEQ     512
#define BATCH   8
#define ROWS    (BATCH*SEQ)   /* 4096 */
#define LN_EPS  1e-5f

// gfx1250 async memory->LDS path (ASYNCcnt), with safe fallback.
#if __has_builtin(__builtin_amdgcn_global_load_async_to_lds_b128) && \
    __has_builtin(__builtin_amdgcn_s_wait_asynccnt)
#define USE_ASYNC_LDS 1
#else
#define USE_ASYNC_LDS 0
#endif

// ---------------------------------------------------------------------------
// LayerNorm: gamma * (x-mean)/(sqrt(var)+eps) + beta   (biased var, std+eps)
// one block per row, 256 threads, D=1024 -> 4 elems/thread. Output bf16.
// ---------------------------------------------------------------------------
__global__ void ln_kernel(const float* __restrict__ x, const float* __restrict__ gamma,
                          const float* __restrict__ beta, __bf16* __restrict__ out) {
  __shared__ float red[256];
  const int row = blockIdx.x;
  const int tid = threadIdx.x;
  const float* xr = x + (size_t)row * D_MODEL;
  float v[4];
  float s = 0.f;
  #pragma unroll
  for (int i = 0; i < 4; ++i) { v[i] = xr[tid + i*256]; s += v[i]; }
  red[tid] = s; __syncthreads();
  for (int off = 128; off > 0; off >>= 1) { if (tid < off) red[tid] += red[tid+off]; __syncthreads(); }
  const float mean = red[0] * (1.0f / D_MODEL);
  __syncthreads();
  float s2 = 0.f;
  #pragma unroll
  for (int i = 0; i < 4; ++i) { float d = v[i]-mean; s2 += d*d; }
  red[tid] = s2; __syncthreads();
  for (int off = 128; off > 0; off >>= 1) { if (tid < off) red[tid] += red[tid+off]; __syncthreads(); }
  const float var = red[0] * (1.0f / D_MODEL);
  const float inv = 1.0f / (sqrtf(var) + LN_EPS);
  #pragma unroll
  for (int i = 0; i < 4; ++i) {
    const int c = tid + i*256;
    out[(size_t)row*D_MODEL + c] = (__bf16)(gamma[c]*((v[i]-mean)*inv) + beta[c]);
  }
}

// ---------------------------------------------------------------------------
// fp32 -> bf16 elementwise convert (weights)
// ---------------------------------------------------------------------------
__global__ void cvt_kernel(const float* __restrict__ in, __bf16* __restrict__ out, int n) {
  const int i = blockIdx.x*256 + threadIdx.x;
  if (i < n) out[i] = (__bf16)in[i];
}

// ---------------------------------------------------------------------------
// Pack Wq/Wk/Wv [H,D,DH] into one bf16 matrix [D, 3*D] with head-major cols:
//   out[d, sect*1024 + h*64 + e] = Wsect[h, d, e]
// ---------------------------------------------------------------------------
__global__ void pack_qkv_kernel(const float* __restrict__ Wq, const float* __restrict__ Wk,
                                const float* __restrict__ Wv, __bf16* __restrict__ out) {
  const int i = blockIdx.x*256 + threadIdx.x;
  if (i >= D_MODEL*3*D_MODEL) return;
  const int d = i / (3*D_MODEL);
  const int c = i % (3*D_MODEL);
  const int sect = c >> 10;
  const int cc   = c & 1023;
  const int h = cc >> 6, e = cc & 63;
  const float* W = (sect == 0) ? Wq : ((sect == 1) ? Wk : Wv);
  out[i] = (__bf16)W[((size_t)h*D_MODEL + d)*DHEAD + e];
}

// ---------------------------------------------------------------------------
// bf16 WMMA GEMM: 64x128 tile/block = A[M,K] * B[K,N] (row-major), fp32 accum.
// 256 threads = 8 wave32; wave (wm,wn): wm 0..3 -> 16-row stripe,
// wn 0..1 -> 64-col half; 4 accumulators per wave (A fragment reused 4x).
// A tile staged straight (async->LDS when available); B tile staged TRANSPOSED
// so each B fragment is one contiguous 32B LDS read.
// EPI: 0 = store bf16 raw
//      2 = fp32 out = residual + acc + bias
//      3 = bf16 out = gelu_exact(acc + bias)
// ---------------------------------------------------------------------------
template<int EPI>
__global__ void gemm_bf16_kernel(const __bf16* __restrict__ A, int lda,
                                 const __bf16* __restrict__ B, int ldb,
                                 const float* __restrict__ bias,
                                 const float* __restrict__ residual,
                                 float* __restrict__ outF,
                                 __bf16* __restrict__ outB,
                                 int ldc, int K) {
  __shared__ __align__(32) __bf16 sA[64*32];     // [m][k]
  __shared__ __align__(32) __bf16 sBt[128*32];   // transposed: [n][k]
  const int tid  = threadIdx.x;
  const int wave = tid >> 5, lane = tid & 31;
  const int wm = wave & 3, wn = wave >> 2;
  const int lm = lane & 15, lh = lane >> 4;
  const int rowBase = blockIdx.y * 64;
  const int colBase = blockIdx.x * 128;
  f32x8 acc[4] = {};
  const int ar = tid >> 2, ac8 = (tid & 3) * 8;  // one 16B chunk of the A tile per thread

  for (int k0 = 0; k0 < K; k0 += 32) {
    __syncthreads();
    // ---- stage A tile 64x32 (straight copy) ----
    {
      const __bf16* gsrc = &A[(size_t)(rowBase + ar)*lda + k0 + ac8];
#if USE_ASYNC_LDS
      __builtin_amdgcn_global_load_async_to_lds_b128(
          (gas_i32x4*)(i32x4*)const_cast<__bf16*>(gsrc),
          (las_i32x4*)(i32x4*)&sA[ar*32 + ac8], 0, 0);
#else
      *(bf16x8*)&sA[ar*32 + ac8] = *(const bf16x8*)gsrc;
#endif
      if (k0 + 32 < K) __builtin_prefetch(gsrc + 32, 0, 3);
    }
    // ---- stage B tile 32x128, transposed into [n][k] ----
    #pragma unroll
    for (int i = 0; i < 16; ++i) {
      const int idx = tid + i*256;
      const int r = idx >> 7, c = idx & 127;          // r = k, c = n
      sBt[c*32 + r] = B[(size_t)(k0 + r)*ldb + colBase + c];
    }
#if USE_ASYNC_LDS
    __builtin_amdgcn_s_wait_asynccnt(0);
#endif
    __syncthreads();

    // A fragment (contiguous k order, same as B: lanes0-15 K=0-15 ; lanes16-31 K=16-31)
    bf16x16 af;
    {
      const __bf16* ap = &sA[(wm*16 + lm)*32];
      const bf16x8 lo = *(const bf16x8*)&ap[lh*16];
      const bf16x8 hi = *(const bf16x8*)&ap[lh*16 + 8];
      af = __builtin_shufflevector(lo, hi, 0,1,2,3,4,5,6,7,8,9,10,11,12,13,14,15);
    }
    #pragma unroll
    for (int nt = 0; nt < 4; ++nt) {
      // B fragment (lanes0-15 N=lane,K=0..15 ; lanes16-31 N=lane-16,K=16..31):
      // contiguous 32B per lane from the transposed tile.
      const int n = wn*64 + nt*16 + lm;
      const bf16x16 bfr = *(const bf16x16*)&sBt[n*32 + lh*16];
      acc[nt] = __builtin_amdgcn_wmma_f32_16x16x32_bf16(
          false, af, false, bfr, (short)0, acc[nt], false, false);
    }
  }

  // Epilogue. C/D layout: VGPR i, lanes0-15 -> M=i, lanes16-31 -> M=8+i; N = lane&15.
  #pragma unroll
  for (int nt = 0; nt < 4; ++nt) {
    #pragma unroll
    for (int i = 0; i < 8; ++i) {
      const int r = rowBase + wm*16 + lh*8 + i;
      const int c = colBase + wn*64 + nt*16 + lm;
      const float a = acc[nt][i];
      if (EPI == 0) {
        outB[(size_t)r*ldc + c] = (__bf16)a;
      } else if (EPI == 2) {
        outF[(size_t)r*ldc + c] = residual[(size_t)r*ldc + c] + a + bias[c];
      } else {
        const float t = a + bias[c];
        const float g = 0.5f*t*(1.0f + erff(t*0.70710678118654752f)); // exact GELU
        outB[(size_t)r*ldc + c] = (__bf16)g;
      }
    }
  }
}

// ---------------------------------------------------------------------------
// Causal attention for one (b, h, 64-query-row block).
// qkv: [4096, 3072] bf16 ; Q cols [h*64,..), K cols [1024+h*64,..), V cols [2048+h*64,..)
// Dynamic LDS: scores 64x512 fp32 (128KB) + transposed V tile 64x32 bf16 (4KB).
// ---------------------------------------------------------------------------
__global__ void attn_kernel(const __bf16* __restrict__ qkv, __bf16* __restrict__ concat) {
  extern __shared__ char smem[];
  float*  sc = (float*)smem;                               // 64 x 512
  __bf16* vt = (__bf16*)(smem + 64*512*sizeof(float));     // transposed [n=64][k=32]
  const int tid  = threadIdx.x;
  const int wave = tid >> 5, lane = tid & 31;
  const int wm = wave & 3, wn = wave >> 2;
  const int lm = lane & 15, lh = lane >> 4;
  const int mblk = blockIdx.x, h = blockIdx.y, b = blockIdx.z;
  const int rowB = b * SEQ;
  const int q0   = mblk * 64;
  const int m    = wm*16 + lm;
  const size_t qoff = (size_t)(rowB + q0 + m)*3072 + h*64;

  // ---- scores = (Q K^T) / 8, causal mask, into LDS ----
  for (int kt = 0; kt < 8; ++kt) {
    f32x8 acc[2] = {};
    #pragma unroll
    for (int ks = 0; ks < 2; ++ks) {        // DH=64 -> 2 k-steps of 32
      bf16x16 af;
      {
        const __bf16* qp = qkv + qoff + ks*32;
        const bf16x8 lo = *(const bf16x8*)&qp[lh*16];
        const bf16x8 hi = *(const bf16x8*)&qp[lh*16 + 8];
        af = __builtin_shufflevector(lo, hi, 0,1,2,3,4,5,6,7,8,9,10,11,12,13,14,15);
      }
      #pragma unroll
      for (int nt = 0; nt < 2; ++nt) {
        const int key = kt*64 + (wn*2+nt)*16 + lm;
        // K rows are contiguous in dh: one 32B global read per lane.
        const bf16x16 bfr = *(const bf16x16*)(qkv + (size_t)(rowB + key)*3072
                                              + 1024 + h*64 + ks*32 + lh*16);
        acc[nt] = __builtin_amdgcn_wmma_f32_16x16x32_bf16(
            false, af, false, bfr, (short)0, acc[nt], false, false);
      }
    }
    #pragma unroll
    for (int nt = 0; nt < 2; ++nt)
      #pragma unroll
      for (int i = 0; i < 8; ++i) {
        const int r = wm*16 + lh*8 + i;
        const int c = kt*64 + (wn*2+nt)*16 + lm;
        sc[r*512 + c] = (c <= q0 + r) ? acc[nt][i]*0.125f : -1e30f;
      }
  }
  __syncthreads();

  // ---- softmax, one thread per query row ----
  if (tid < 64) {
    float* rowp = sc + tid*512;
    float mx = -1e30f;
    for (int j = 0; j < 512; ++j) mx = fmaxf(mx, rowp[j]);
    float sum = 0.f;
    for (int j = 0; j < 512; ++j) { const float e = __expf(rowp[j]-mx); rowp[j] = e; sum += e; }
    const float inv = 1.0f/sum;
    for (int j = 0; j < 512; ++j) rowp[j] *= inv;
  }

  // ---- ctx = P @ V ----
  f32x8 acc2[2] = {};
  for (int ks = 0; ks < 16; ++ks) {          // K=512 -> 16 k-steps of 32
    __syncthreads();
    #pragma unroll
    for (int i = 0; i < 8; ++i) {            // stage V tile 32x64, transposed [n][k]
      const int idx = tid + i*256;
      const int r = idx >> 6, c = idx & 63;  // r = k, c = n
      vt[c*32 + r] = qkv[(size_t)(rowB + ks*32 + r)*3072 + 2048 + h*64 + c];
    }
    __syncthreads();
    bf16x16 af;
    {
      const float* pp = sc + m*512 + ks*32;
      const f32x8 plo = *(const f32x8*)&pp[lh*16];
      const f32x8 phi = *(const f32x8*)&pp[lh*16 + 8];
      #pragma unroll
      for (int e = 0; e < 8; ++e) { af[e] = (__bf16)plo[e]; af[8+e] = (__bf16)phi[e]; }
    }
    #pragma unroll
    for (int nt = 0; nt < 2; ++nt) {
      const int n = (wn*2+nt)*16 + lm;
      const bf16x16 bfr = *(const bf16x16*)&vt[n*32 + lh*16];
      acc2[nt] = __builtin_amdgcn_wmma_f32_16x16x32_bf16(
          false, af, false, bfr, (short)0, acc2[nt], false, false);
    }
  }
  #pragma unroll
  for (int nt = 0; nt < 2; ++nt)
    #pragma unroll
    for (int i = 0; i < 8; ++i) {
      const int r = wm*16 + lh*8 + i;
      const int c = (wn*2+nt)*16 + lm;
      concat[(size_t)(rowB + q0 + r)*D_MODEL + h*64 + c] = (__bf16)acc2[nt][i];
    }
}

// ---------------------------------------------------------------------------
extern "C" void kernel_launch(void* const* d_in, const int* in_sizes, int n_in,
                              void* d_out, int out_size, void* d_ws, size_t ws_size,
                              hipStream_t stream) {
  (void)in_sizes; (void)n_in; (void)out_size; (void)ws_size;
  const float* x   = (const float*)d_in[0];
  const float* Wq  = (const float*)d_in[1];
  const float* Wk  = (const float*)d_in[2];
  const float* Wv  = (const float*)d_in[3];
  const float* Wo  = (const float*)d_in[4];
  const float* bo  = (const float*)d_in[5];
  const float* W1  = (const float*)d_in[6];
  const float* b1  = (const float*)d_in[7];
  const float* W2  = (const float*)d_in[8];
  const float* b2  = (const float*)d_in[9];
  const float* g1  = (const float*)d_in[10];
  const float* be1 = (const float*)d_in[11];
  const float* g2  = (const float*)d_in[12];
  const float* be2 = (const float*)d_in[13];

  char* ws = (char*)d_ws;
  size_t off = 0;
  __bf16* xn     = (__bf16*)(ws + off); off += (size_t)ROWS*D_MODEL*2;        // 8 MB
  __bf16* wqkv   = (__bf16*)(ws + off); off += (size_t)D_MODEL*3*D_MODEL*2;   // 6 MB
  __bf16* wo     = (__bf16*)(ws + off); off += (size_t)D_MODEL*D_MODEL*2;     // 2 MB
  __bf16* w1     = (__bf16*)(ws + off); off += (size_t)D_MODEL*FFDIM*2;       // 8 MB
  __bf16* w2     = (__bf16*)(ws + off); off += (size_t)FFDIM*D_MODEL*2;       // 8 MB
  __bf16* qkvb   = (__bf16*)(ws + off); off += (size_t)ROWS*3*D_MODEL*2;      // 24 MB
  __bf16* concat = (__bf16*)(ws + off); off += (size_t)ROWS*D_MODEL*2;        // 8 MB
  float*  out1   = (float*)(ws + off);  off += (size_t)ROWS*D_MODEL*4;        // 16 MB
  __bf16* xn2    = (__bf16*)(ws + off); off += (size_t)ROWS*D_MODEL*2;        // 8 MB
  __bf16* hb     = (__bf16*)(ws + off); off += (size_t)ROWS*FFDIM*2;          // 32 MB

  const dim3 blk(256);

  // 1. LN1 -> bf16
  ln_kernel<<<ROWS, blk, 0, stream>>>(x, g1, be1, xn);
  // 2. weight conversions
  pack_qkv_kernel<<<(D_MODEL*3*D_MODEL + 255)/256, blk, 0, stream>>>(Wq, Wk, Wv, wqkv);
  cvt_kernel<<<(D_MODEL*D_MODEL + 255)/256, blk, 0, stream>>>(Wo, wo, D_MODEL*D_MODEL);
  cvt_kernel<<<(D_MODEL*FFDIM  + 255)/256, blk, 0, stream>>>(W1, w1, D_MODEL*FFDIM);
  cvt_kernel<<<(FFDIM*D_MODEL  + 255)/256, blk, 0, stream>>>(W2, w2, FFDIM*D_MODEL);
  // 3. fused QKV GEMM (4096 x 3072 x 1024) -> bf16
  gemm_bf16_kernel<0><<<dim3(3*D_MODEL/128, ROWS/64), blk, 0, stream>>>(
      xn, D_MODEL, wqkv, 3*D_MODEL, nullptr, nullptr, nullptr, qkvb, 3*D_MODEL, D_MODEL);
  // 4. causal attention -> bf16 concat
  attn_kernel<<<dim3(SEQ/64, NHEAD, BATCH), blk, 64*512*4 + 64*32*2, stream>>>(qkvb, concat);
  // 5. out-proj + bias + residual(x) -> fp32 out1
  gemm_bf16_kernel<2><<<dim3(D_MODEL/128, ROWS/64), blk, 0, stream>>>(
      concat, D_MODEL, wo, D_MODEL, bo, x, out1, nullptr, D_MODEL, D_MODEL);
  // 6. LN2 -> bf16
  ln_kernel<<<ROWS, blk, 0, stream>>>(out1, g2, be2, xn2);
  // 7. FFN1 + bias + exact GELU -> bf16
  gemm_bf16_kernel<3><<<dim3(FFDIM/128, ROWS/64), blk, 0, stream>>>(
      xn2, D_MODEL, w1, FFDIM, b1, nullptr, nullptr, hb, FFDIM, D_MODEL);
  // 8. FFN2 + bias + residual(out1) -> fp32 d_out
  gemm_bf16_kernel<2><<<dim3(D_MODEL/128, ROWS/64), blk, 0, stream>>>(
      hb, FFDIM, w2, D_MODEL, b2, out1, (float*)d_out, nullptr, D_MODEL, FFDIM);
}
